// EncoderRNN_9285719294595
// MI455X (gfx1250) — matrix operand order repaired and checked by reading, not compiled, not run on hardware
//
#include <hip/hip_runtime.h>
#include <hip/hip_bf16.h>

// Problem dims (compile-time constants from the reference)
#define Bsz 64
#define Tn  1024
#define In  64
#define Hn  256
#define G3  768          // 3*H
#define PAIRS (Hn / 2)   // 128 bf16 k-pairs
#define HSTR  (Bsz + 1)  // padded dword stride per k-pair row (bank-conflict-free)

typedef __attribute__((ext_vector_type(16))) __bf16 v16bf;
typedef __attribute__((ext_vector_type(8)))  float  v8f;

union FragAB { v16bf v; unsigned int d[8]; };

#define Z8 {0.f,0.f,0.f,0.f,0.f,0.f,0.f,0.f}

__device__ __forceinline__ unsigned int f2bf(float f) {
    unsigned int u = __float_as_uint(f);
    return ((u + 0x7FFFu + ((u >> 16) & 1u)) >> 16) & 0xFFFFu; // RNE
}

__device__ __forceinline__ float sigmoidf_fast(float x) {
    return 1.0f / (1.0f + __expf(-x));
}

// ---- WMMA fragment loaders -------------------------------------------------
// 16-bit A layout (16x32, MxK): lane L -> row M=L%16, half=L/16.
//   dword j in 0..3: K = 2j + 8*half ; j in 4..7: K = 16 + 2(j-4) + 8*half
__device__ __forceinline__ int kpair(int j, int halfk) {
    return (j < 4) ? (j * 2 + halfk * 8) : (16 + (j - 4) * 2 + halfk * 8);
}

// A fragment from a global bf16 row (row base for this lane's M row)
__device__ __forceinline__ FragAB loadA_g(const unsigned short* __restrict__ arow,
                                          int ks, int halfk) {
    FragAB f;
#pragma unroll
    for (int j = 0; j < 8; ++j) {
        const int k0 = ks * 32 + kpair(j, halfk);
        f.d[j] = *(const unsigned int*)(arow + k0);   // k0 even -> dword aligned
    }
    return f;
}

// A fragment from LDS bf16 h, stored as packed pairs: dword[(k>>1)*HSTR + m]
__device__ __forceinline__ FragAB loadA_h(const unsigned int* __restrict__ hbf,
                                          int m, int ks, int halfk) {
    FragAB f;
#pragma unroll
    for (int j = 0; j < 8; ++j) {
        const int k0 = ks * 32 + kpair(j, halfk);     // even
        f.d[j] = hbf[(k0 >> 1) * HSTR + m];
    }
    return f;
}

// B fragment (KxN): lane L -> col N=L%16, kbase=(L/16)*8; dword j packs K=kbase+2j,+1 (j<4), kbase+16+2(j-4),+1 (j>=4).
// Weight W stored row-major [n][k] (bf16), so B[k][n] = W[n][k]: same pair packing.
__device__ __forceinline__ FragAB loadB(const unsigned short* __restrict__ w,
                                        int stride, int nrow, int kb) {
    FragAB f;
    const unsigned short* base = w + nrow * stride + kb;
#pragma unroll
    for (int j = 0; j < 8; ++j) {
        f.d[j] = *(const unsigned int*)(base + ((j < 4) ? 2 * j : 16 + 2 * (j - 4)));
    }
    return f;
}

__device__ __forceinline__ v8f wmma_bf16(const FragAB& a, const FragAB& b, v8f c) {
    return __builtin_amdgcn_wmma_f32_16x16x32_bf16(
        /*neg_a=*/false, a.v, /*neg_b=*/false, b.v,
        /*c_mod=*/(short)0, c, /*reuse_a=*/false, /*reuse_b=*/false);
}

// ---- grid-wide barrier (4 resident workgroups, generation counter) ---------
__device__ __forceinline__ void grid_barrier(unsigned* cnt, unsigned* gen, unsigned nwg) {
    __threadfence();
    __syncthreads();
    if (threadIdx.x == 0) {
        const unsigned g = __hip_atomic_load(gen, __ATOMIC_ACQUIRE, __HIP_MEMORY_SCOPE_AGENT);
        const unsigned a = __hip_atomic_fetch_add(cnt, 1u, __ATOMIC_ACQ_REL, __HIP_MEMORY_SCOPE_AGENT);
        if (a == nwg - 1u) {
            __hip_atomic_store(cnt, 0u, __ATOMIC_RELEASE, __HIP_MEMORY_SCOPE_AGENT);
            __hip_atomic_fetch_add(gen, 1u, __ATOMIC_RELEASE, __HIP_MEMORY_SCOPE_AGENT);
        } else {
            while (__hip_atomic_load(gen, __ATOMIC_ACQUIRE, __HIP_MEMORY_SCOPE_AGENT) == g)
                __builtin_amdgcn_s_sleep(2);
        }
    }
    __syncthreads();
}

// ---- setup kernels ---------------------------------------------------------
__global__ void cvt_f32_to_bf16(const float* __restrict__ src,
                                unsigned short* __restrict__ dst, int n) {
    for (int i = blockIdx.x * blockDim.x + threadIdx.x; i < n; i += gridDim.x * blockDim.x)
        dst[i] = (unsigned short)f2bf(src[i]);
}

__global__ void reset_barrier(unsigned* p) { p[0] = 0u; p[1] = 0u; }

// ---- persistent GRU kernel: 4 WGs, one per (layer, direction) --------------
__global__ __launch_bounds__(512, 1) void gru_persistent(
    const unsigned short* __restrict__ xbf,   // [B][T][I] bf16
    const unsigned short* __restrict__ wih0,  // [2][768][64]  bf16
    const unsigned short* __restrict__ whh0,  // [2][768][256] bf16
    const unsigned short* __restrict__ wih1,  // [2][768][512] bf16
    const unsigned short* __restrict__ whh1,  // [2][768][256] bf16
    unsigned short* __restrict__ y0,          // [2][B][2H] bf16 double buffer
    const float* __restrict__ bih0, const float* __restrict__ bhh0,
    const float* __restrict__ bih1, const float* __restrict__ bhh1,
    const int* __restrict__ seqlen,
    float* __restrict__ out,                  // [1][B][2H] fp32
    unsigned* __restrict__ bar)
{
    // bf16 GEMM copy of h: packed pairs, padded stride (33 KB). fp32 carry lives
    // in registers (hprev) since the writer thread of h(m,j) is also its reader.
    __shared__ unsigned int hbf[PAIRS * HSTR];

    const int unit  = blockIdx.x;             // 0..3
    const int layer = unit >> 1;
    const int dir   = unit & 1;
    const int lane  = threadIdx.x & 31;
    const int wave  = threadIdx.x >> 5;       // 0..15
    const int halfk = lane >> 4;
    const int lcol  = lane & 15;

    for (int i = threadIdx.x; i < PAIRS * HSTR; i += 512) hbf[i] = 0u;

    const unsigned short* Wih = layer ? (wih1 + dir * G3 * (2 * Hn)) : (wih0 + dir * G3 * In);
    const int Kin = layer ? (2 * Hn) : In;
    const unsigned short* Whh = (layer ? whh1 : whh0) + dir * G3 * Hn;
    const float* bih = (layer ? bih1 : bih0) + dir * G3;
    const float* bhh = (layer ? bhh1 : bhh0) + dir * G3;

    // Loop-invariant per-tile bias terms (hoisted across all 1024 steps)
    float b_r[4], b_z[4], b_in[4], b_hn[4];
#pragma unroll
    for (int tu = 0; tu < 4; ++tu) {
        const int tid = wave * 4 + tu;
        const int nr  = (tid & 15) * 16 + lcol;
        b_r[tu]  = bih[nr] + bhh[nr];
        b_z[tu]  = bih[Hn + nr] + bhh[Hn + nr];
        b_in[tu] = bih[2 * Hn + nr];
        b_hn[tu] = bhh[2 * Hn + nr];
    }

    // fp32 hidden-state carry in registers: hprev[tu][e] <-> h(mr, nr)
    float hprev[4][8];
#pragma unroll
    for (int tu = 0; tu < 4; ++tu)
#pragma unroll
        for (int e = 0; e < 8; ++e) hprev[tu][e] = 0.f;

    __syncthreads();

    // Software pipeline: L0 works on step t=it, L1 on t=it-1 (consumes y0[t&1]).
    for (int it = 0; it <= Tn; ++it) {
        const int  t      = layer ? (it - 1) : it;
        const bool active = layer ? (it >= 1) : (it < Tn);

        if (active) {
#pragma unroll
            for (int tu = 0; tu < 4; ++tu) {
                const int tid = wave * 4 + tu;     // 0..63 tile id
                const int mt  = tid >> 4;          // 0..3   (batch tile)
                const int jt  = tid & 15;          // 0..15  (hidden tile)
                const int m   = mt * 16 + lcol;    // this lane's A row (batch)
                const int nr  = jt * 16 + lcol;    // this lane's B row within H

                v8f gir = Z8, giz = Z8, gin = Z8, ghr = Z8, ghz = Z8, ghn = Z8;

                // ---- input-path GEMM: Gi = X * W_ih^T ----
                const unsigned short* arow = layer
                    ? (y0 + (((t & 1) * Bsz + m) * (2 * Hn)))
                    : (xbf + ((m * Tn + t) * In));
                for (int ks = 0; ks < (Kin >> 5); ++ks) {
                    const FragAB a  = loadA_g(arow, ks, halfk);
                    const int    kb = ks * 32 + halfk * 8;
                    const FragAB br = loadB(Wih, Kin, 0 * Hn + nr, kb);
                    const FragAB bz = loadB(Wih, Kin, 1 * Hn + nr, kb);
                    const FragAB bn = loadB(Wih, Kin, 2 * Hn + nr, kb);
                    gir = wmma_bf16(a, br, gir);
                    giz = wmma_bf16(a, bz, giz);
                    gin = wmma_bf16(a, bn, gin);
                }

                // ---- hidden-path GEMM: Gh = h * W_hh^T (A from LDS bf16) ----
                for (int ks = 0; ks < (Hn >> 5); ++ks) {
                    const FragAB a  = loadA_h(hbf, m, ks, halfk);
                    const int    kb = ks * 32 + halfk * 8;
                    const FragAB br = loadB(Whh, Hn, 0 * Hn + nr, kb);
                    const FragAB bz = loadB(Whh, Hn, 1 * Hn + nr, kb);
                    const FragAB bn = loadB(Whh, Hn, 2 * Hn + nr, kb);
                    ghr = wmma_bf16(a, br, ghr);
                    ghz = wmma_bf16(a, bz, ghz);
                    ghn = wmma_bf16(a, bn, ghn);
                }

                // ---- GRU nonlinearity; fp32 carry stays in registers ----
                // C/D layout: VGPR e -> M = mt*16 + 8*halfk + e, lane -> N = nr
#pragma unroll
                for (int e = 0; e < 8; ++e) {
                    const float r  = sigmoidf_fast(gir[e] + ghr[e] + b_r[tu]);
                    const float z  = sigmoidf_fast(giz[e] + ghz[e] + b_z[tu]);
                    const float nn = tanhf(gin[e] + b_in[tu] + r * (ghn[e] + b_hn[tu]));
                    hprev[tu][e] = (1.f - z) * nn + z * hprev[tu][e];
                }
            }
        }

        __syncthreads();   // all waves finished reading old bf16 h from LDS

        if (active) {
            unsigned short* hb16 = (unsigned short*)hbf;
#pragma unroll
            for (int tu = 0; tu < 4; ++tu) {
                const int tid = wave * 4 + tu;
                const int mt  = tid >> 4;
                const int jt  = tid & 15;
                const int nr  = jt * 16 + lcol;
#pragma unroll
                for (int e = 0; e < 8; ++e) {
                    const int   mr = mt * 16 + halfk * 8 + e;
                    const float hn = hprev[tu][e];
                    // publish bf16 h for next step's Gh GEMM (ds_store_b16)
                    hb16[((nr >> 1) * HSTR + mr) * 2 + (nr & 1)] = (unsigned short)f2bf(hn);
                    if (layer == 0) {
                        // publish concat(h_d0, h_d1) for layer 1, double-buffered
                        y0[(((t & 1) * Bsz + mr) * (2 * Hn)) + dir * Hn + nr] =
                            (unsigned short)f2bf(hn);
                    } else {
                        int sl = seqlen[mr] - 1;
                        sl = sl < 0 ? 0 : (sl > Tn - 1 ? Tn - 1 : sl);
                        if (sl == t)
                            out[mr * (2 * Hn) + dir * Hn + nr] = hn;
                    }
                }
            }
        }

        grid_barrier(bar, bar + 1, 4u);
    }
}

// ---- host launcher ---------------------------------------------------------
extern "C" void kernel_launch(void* const* d_in, const int* in_sizes, int n_in,
                              void* d_out, int out_size, void* d_ws, size_t ws_size,
                              hipStream_t stream) {
    const float* x    = (const float*)d_in[0];  // [64,1024,64]
    const float* Wih0 = (const float*)d_in[1];  // [2,768,64]
    const float* Whh0 = (const float*)d_in[2];  // [2,768,256]
    const float* bih0 = (const float*)d_in[3];  // [2,768]
    const float* bhh0 = (const float*)d_in[4];  // [2,768]
    const float* Wih1 = (const float*)d_in[5];  // [2,768,512]
    const float* Whh1 = (const float*)d_in[6];  // [2,768,256]
    const float* bih1 = (const float*)d_in[7];  // [2,768]
    const float* bhh1 = (const float*)d_in[8];  // [2,768]
    const int*   seq  = (const int*)d_in[9];    // [64]
    float*       out  = (float*)d_out;          // [1,64,512]

    char* wsb = (char*)d_ws;
    unsigned* bar = (unsigned*)wsb;                       // 2 uints (cnt, gen)
    unsigned short* hbase = (unsigned short*)(wsb + 256); // bf16 region

    const int nWih0 = 2 * G3 * In;        // 98304
    const int nWhh0 = 2 * G3 * Hn;        // 393216
    const int nWih1 = 2 * G3 * (2 * Hn);  // 786432
    const int nWhh1 = 2 * G3 * Hn;        // 393216
    const int nX    = Bsz * Tn * In;      // 4194304

    unsigned short* wih0b = hbase;
    unsigned short* whh0b = wih0b + nWih0;
    unsigned short* wih1b = whh0b + nWhh0;
    unsigned short* whh1b = wih1b + nWih1;
    unsigned short* xbf   = whh1b + nWhh1;
    unsigned short* y0    = xbf + nX;     // [2][64][512] bf16 double buffer

    reset_barrier<<<1, 1, 0, stream>>>(bar);
    cvt_f32_to_bf16<<<256, 256, 0, stream>>>(Wih0, wih0b, nWih0);
    cvt_f32_to_bf16<<<512, 256, 0, stream>>>(Whh0, whh0b, nWhh0);
    cvt_f32_to_bf16<<<512, 256, 0, stream>>>(Wih1, wih1b, nWih1);
    cvt_f32_to_bf16<<<512, 256, 0, stream>>>(Whh1, whh1b, nWhh1);
    cvt_f32_to_bf16<<<1024, 256, 0, stream>>>(x, xbf, nX);

    gru_persistent<<<4, 512, 0, stream>>>(
        xbf, wih0b, whh0b, wih1b, whh1b, y0,
        bih0, bhh0, bih1, bhh1, seq, out, bar);

    (void)in_sizes; (void)n_in; (void)out_size; (void)ws_size;
}
